// AttentionPairBias_86371792322643
// MI455X (gfx1250) — matrix operand order repaired and checked by reading, not compiled, not run on hardware
//
#include <hip/hip_runtime.h>

typedef __attribute__((ext_vector_type(16))) _Float16 v16h;
typedef __attribute__((ext_vector_type(8)))  float    v8f;
typedef __attribute__((ext_vector_type(4)))  unsigned int u32x4;

#define NSEQ 1024
#define CA   384
#define CZ   128
#define NH   16
#define CH   24

union V16 { v16h v; u32x4 q[2]; _Float16 h[16]; };

__device__ __forceinline__ v8f wmma_f16(v16h a, v16h b, v8f c) {
  return __builtin_amdgcn_wmma_f32_16x16x32_f16(false, a, false, b, (short)0, c, false, false);
}

__device__ __forceinline__ float sigmoidf_(float x) { return 1.0f / (1.0f + __expf(-x)); }

// A fragment (16x32 f16, M x K): lane m = lane&15, chunks at K = hi*8 and 16+hi*8.
// base points at element (row0, k0) of a row-major [*, ld] f16 matrix.
__device__ __forceinline__ v16h load_a_frag(const _Float16* base, int ld) {
  int lane = threadIdx.x & 31;
  int m = lane & 15, hi = lane >> 4;
  const _Float16* p = base + m * ld + hi * 8;
  V16 u;
  u.q[0] = *(const u32x4*)(p);
  u.q[1] = *(const u32x4*)(p + 16);
  return u.v;
}

// B fragment (32x16 f16, K x N) for X @ W^T with W row-major [Cout, K]:
// B[kk][nn] = W[c0+nn][k0+kk]. Pass base = W + c0*ld + k0.
// lane n = lane&15, chunks at K = hi*8 and 16+hi*8.
__device__ __forceinline__ v16h load_b_fragT(const _Float16* base, int ld) {
  int lane = threadIdx.x & 31;
  int n = lane & 15, hi = lane >> 4;
  const _Float16* p = base + n * ld + hi * 8;
  V16 u;
  u.q[0] = *(const u32x4*)(p);
  u.q[1] = *(const u32x4*)(p + 16);
  return u.v;
}

// ---------------------------------------------------------------- K1: weights -> f16
__global__ __launch_bounds__(256) void cvt_weights_kernel(
    const float* w0, const float* w1, const float* w2, const float* w3,
    const float* w4, const float* w5, const float* w6, const float* w7,
    _Float16* o0, _Float16* o1, _Float16* o2, _Float16* o3,
    _Float16* o4, _Float16* o5, _Float16* o6, _Float16* o7) {
  int i = blockIdx.x * blockDim.x + threadIdx.x;
  if (i < CA * CA) {
    o0[i] = (_Float16)w0[i]; o1[i] = (_Float16)w1[i];
    o2[i] = (_Float16)w2[i]; o3[i] = (_Float16)w3[i];
    o4[i] = (_Float16)w4[i]; o5[i] = (_Float16)w5[i];
    o6[i] = (_Float16)w6[i]; o7[i] = (_Float16)w7[i];
  }
}

// ---------------------------------------------------------------- K2: row LayerNorms
__device__ __forceinline__ float block_sum_384(float v, float* red, int t) {
  __syncthreads();
  red[t] = v; __syncthreads();
  if (t < 192) red[t] += red[t + 192]; __syncthreads();
  if (t <  96) red[t] += red[t +  96]; __syncthreads();
  if (t <  48) red[t] += red[t +  48]; __syncthreads();
  if (t <  24) red[t] += red[t +  24]; __syncthreads();
  if (t <  12) red[t] += red[t +  12]; __syncthreads();
  if (t <   6) red[t] += red[t +   6]; __syncthreads();
  if (t <   3) red[t] += red[t +   3]; __syncthreads();
  return red[0] + red[1] + red[2];
}

__global__ __launch_bounds__(384) void rowln_kernel(
    const float* __restrict__ a, const float* __restrict__ s, const float* __restrict__ ln_s_w,
    float* __restrict__ a_n, _Float16* __restrict__ s_n_h, _Float16* __restrict__ s_h) {
  __shared__ float red[384];
  int row = blockIdx.x, t = threadIdx.x;
  float va = a[row * CA + t];
  float vs = s[row * CA + t];
  float asum = block_sum_384(va, red, t);
  float asq  = block_sum_384(va * va, red, t);
  float ssum = block_sum_384(vs, red, t);
  float ssq  = block_sum_384(vs * vs, red, t);
  float am = asum * (1.0f / CA), av = asq * (1.0f / CA) - am * am;
  float sm = ssum * (1.0f / CA), sv = ssq * (1.0f / CA) - sm * sm;
  float an = (va - am) * rsqrtf(av + 1e-5f);
  float sn = (vs - sm) * rsqrtf(sv + 1e-5f) * ln_s_w[t];
  a_n[row * CA + t]   = an;
  s_n_h[row * CA + t] = (_Float16)sn;
  s_h[row * CA + t]   = (_Float16)vs;
}

// ---------------------------------------------------------------- K3: AdaLN GEMMs
__global__ __launch_bounds__(128) void adaln_kernel(
    const _Float16* __restrict__ s_n_h, const float* __restrict__ a_n,
    const _Float16* __restrict__ wgam, const _Float16* __restrict__ wbet,
    const float* __restrict__ b_gamma, _Float16* __restrict__ a_c_h) {
  int wave = threadIdx.x >> 5, lane = threadIdx.x & 31;
  int tid = blockIdx.x * 4 + wave;
  int i0 = (tid / 24) * 16, c0 = (tid % 24) * 16;
  int n = lane & 15, hi = lane >> 4;
  v8f accG = {0.f,0.f,0.f,0.f,0.f,0.f,0.f,0.f};
  v8f accB = {0.f,0.f,0.f,0.f,0.f,0.f,0.f,0.f};
  for (int k0 = 0; k0 < CA; k0 += 32) {
    v16h af = load_a_frag(s_n_h + i0 * CA + k0, CA);
    accG = wmma_f16(af, load_b_fragT(wgam + c0 * CA + k0, CA), accG);
    accB = wmma_f16(af, load_b_fragT(wbet + c0 * CA + k0, CA), accB);
  }
  float bg = b_gamma[c0 + n];
#pragma unroll
  for (int r = 0; r < 8; ++r) {
    int m = hi * 8 + r;
    float sig = sigmoidf_(accG[r] + bg);
    float val = a_n[(i0 + m) * CA + c0 + n] * sig + accB[r];
    a_c_h[(i0 + m) * CA + c0 + n] = (_Float16)val;
  }
}

// ---------------------------------------------------------------- K4: Q/K/V/G projections
__global__ __launch_bounds__(128) void qkvg_kernel(
    const _Float16* __restrict__ a_c_h,
    const _Float16* __restrict__ wq, const _Float16* __restrict__ wk,
    const _Float16* __restrict__ wv, const _Float16* __restrict__ wg,
    const float* __restrict__ bq,
    _Float16* __restrict__ q_h, _Float16* __restrict__ k_h,
    _Float16* __restrict__ v_t, float* __restrict__ g_f) {
  int wave = threadIdx.x >> 5, lane = threadIdx.x & 31;
  int tid = blockIdx.x * 4 + wave;
  int i0 = (tid / 24) * 16, c0 = (tid % 24) * 16;
  int n = lane & 15, hi = lane >> 4;
  v8f aq = {0.f,0.f,0.f,0.f,0.f,0.f,0.f,0.f};
  v8f ak = aq, av = aq, ag = aq;
  for (int k0 = 0; k0 < CA; k0 += 32) {
    v16h af = load_a_frag(a_c_h + i0 * CA + k0, CA);
    aq = wmma_f16(af, load_b_fragT(wq + c0 * CA + k0, CA), aq);
    ak = wmma_f16(af, load_b_fragT(wk + c0 * CA + k0, CA), ak);
    av = wmma_f16(af, load_b_fragT(wv + c0 * CA + k0, CA), av);
    ag = wmma_f16(af, load_b_fragT(wg + c0 * CA + k0, CA), ag);
  }
  const float rscale = 0.2041241452319315f;  // 1/sqrt(24)
  float b = bq[c0 + n];
#pragma unroll
  for (int r = 0; r < 8; ++r) {
    int m = hi * 8 + r;
    int idx = (i0 + m) * CA + c0 + n;
    q_h[idx] = (_Float16)((aq[r] + b) * rscale);
    k_h[idx] = (_Float16)ak[r];
    v_t[(c0 + n) * NSEQ + (i0 + m)] = (_Float16)av[r];  // transposed [H*CH][N]
    g_f[idx] = sigmoidf_(ag[r]);
  }
}

// ---------------------------------------------------------------- K5: pair bias (streams z once)
__global__ __launch_bounds__(256) void zbias_kernel(
    const float* __restrict__ z, const float* __restrict__ lnz_w, const float* __restrict__ lnz_b,
    const float* __restrict__ wz, _Float16* __restrict__ zb) {
  int wave = threadIdx.x >> 5, lane = threadIdx.x & 31;
  int pid = blockIdx.x * 8 + wave;               // pid = i*1024 + j
  const float* zp = z + (long)pid * CZ + lane * 4;
  float4 zv = *(const float4*)zp;
  float sum = zv.x + zv.y + zv.z + zv.w;
  float sq  = zv.x * zv.x + zv.y * zv.y + zv.z * zv.z + zv.w * zv.w;
#pragma unroll
  for (int off = 16; off > 0; off >>= 1) {
    sum += __shfl_xor(sum, off, 32);
    sq  += __shfl_xor(sq,  off, 32);
  }
  float mean = sum * (1.0f / CZ);
  float var  = sq  * (1.0f / CZ) - mean * mean;
  float rinv = rsqrtf(var + 1e-5f);
  float4 lw = *(const float4*)(lnz_w + lane * 4);
  float4 lb = *(const float4*)(lnz_b + lane * 4);
  float z0 = (zv.x - mean) * rinv * lw.x + lb.x;
  float z1 = (zv.y - mean) * rinv * lw.y + lb.y;
  float z2 = (zv.z - mean) * rinv * lw.z + lb.z;
  float z3 = (zv.w - mean) * rinv * lw.w + lb.w;
  float acc[NH];
#pragma unroll
  for (int h = 0; h < NH; ++h) {
    float4 w = *(const float4*)(wz + h * CZ + lane * 4);
    acc[h] = z0 * w.x + z1 * w.y + z2 * w.z + z3 * w.w;
  }
#pragma unroll
  for (int off = 16; off > 0; off >>= 1) {
#pragma unroll
    for (int h = 0; h < NH; ++h) acc[h] += __shfl_xor(acc[h], off, 32);
  }
  if (lane < NH) zb[(lane << 20) + pid] = (_Float16)acc[lane];
}

// ---------------------------------------------------------------- K6: flash attention
__global__ __launch_bounds__(128) void attn_kernel(
    const _Float16* __restrict__ q_h, const _Float16* __restrict__ k_h,
    const _Float16* __restrict__ v_t, const _Float16* __restrict__ zb,
    const float* __restrict__ mask, const float* __restrict__ g_f,
    _Float16* __restrict__ og) {
  __shared__ _Float16 plds[4][16 * 32];
  int wave = threadIdx.x >> 5, lane = threadIdx.x & 31;
  int tid = blockIdx.x * 4 + wave;
  int h = tid >> 6, i0 = (tid & 63) << 4;
  int n = lane & 15, hi = lane >> 4;
  const u32x4 z4 = {0u, 0u, 0u, 0u};

  // Q A-fragment, K padded 24->32 (zero chunk with K>=24)
  V16 qf;
  const _Float16* qp = q_h + (i0 + n) * CA + h * CH + hi * 8;
  qf.q[0] = *(const u32x4*)qp;
  if (hi) qf.q[1] = z4; else qf.q[1] = *(const u32x4*)(qp + 16);

  v8f o0 = {0.f,0.f,0.f,0.f,0.f,0.f,0.f,0.f};
  v8f o1 = o0;
  float mrun[8], lrun[8];
#pragma unroll
  for (int r = 0; r < 8; ++r) { mrun[r] = -3.0e38f; lrun[r] = 0.0f; }
  const v8f zc = {0.f,0.f,0.f,0.f,0.f,0.f,0.f,0.f};
  _Float16* L = plds[wave];

  for (int j0 = 0; j0 < NSEQ; j0 += 32) {
    if (j0 + 32 < NSEQ) __builtin_prefetch(k_h + (j0 + 32 + n) * CA + h * CH, 0, 1);
    // K B-fragments for two 16-wide key tiles (pad K>=24 with zeros)
    V16 kf0, kf1;
    {
      const _Float16* p = k_h + (j0 + n) * CA + h * CH + hi * 8;
      kf0.q[0] = *(const u32x4*)p;
      kf0.q[1] = hi ? z4 : *(const u32x4*)(p + 16);
      const _Float16* p2 = p + 16 * CA;
      kf1.q[0] = *(const u32x4*)p2;
      kf1.q[1] = hi ? z4 : *(const u32x4*)(p2 + 16);
    }
    v8f lg0 = wmma_f16(qf.v, kf0.v, zc);
    v8f lg1 = wmma_f16(qf.v, kf1.v, zc);

    float mb0 = (mask[j0 + n] - 1.0f) * 1e9f;
    float mb1 = (mask[j0 + 16 + n] - 1.0f) * 1e9f;
    float t0[8], t1[8], mx[8];
#pragma unroll
    for (int r = 0; r < 8; ++r) {
      int m = hi * 8 + r;
      const _Float16* zp = zb + (h << 20) + ((i0 + m) << 10);
      t0[r] = lg0[r] + (float)zp[j0 + n] + mb0;
      t1[r] = lg1[r] + (float)zp[j0 + 16 + n] + mb1;
      mx[r] = fmaxf(t0[r], t1[r]);
    }
#pragma unroll
    for (int off = 8; off > 0; off >>= 1)
#pragma unroll
      for (int r = 0; r < 8; ++r) mx[r] = fmaxf(mx[r], __shfl_xor(mx[r], off, 32));
    float p0[8], p1[8], rs[8];
#pragma unroll
    for (int r = 0; r < 8; ++r) {
      float mn = fmaxf(mrun[r], mx[r]);
      float sc = __expf(mrun[r] - mn);
      mrun[r] = mn;
      p0[r] = __expf(t0[r] - mn);
      p1[r] = __expf(t1[r] - mn);
      rs[r] = p0[r] + p1[r];
      lrun[r] *= sc; o0[r] *= sc; o1[r] *= sc;
    }
#pragma unroll
    for (int off = 8; off > 0; off >>= 1)
#pragma unroll
      for (int r = 0; r < 8; ++r) rs[r] += __shfl_xor(rs[r], off, 32);
#pragma unroll
    for (int r = 0; r < 8; ++r) lrun[r] += rs[r];

    // stage P (D-layout) -> LDS -> A-layout fragment
#pragma unroll
    for (int r = 0; r < 8; ++r) {
      int m = hi * 8 + r;
      L[m * 32 + n]      = (_Float16)p0[r];
      L[m * 32 + 16 + n] = (_Float16)p1[r];
    }
    V16 pf;
    const _Float16* pp = L + n * 32 + hi * 16;
    pf.q[0] = *(const u32x4*)pp;
    pf.q[1] = *(const u32x4*)(pp + 8);

    // V B-fragments from transposed v_t (contiguous in j)
    V16 vf0, vf1;
    {
      const _Float16* vp = v_t + (h * CH + n) * NSEQ + j0 + hi * 16;
      vf0.q[0] = *(const u32x4*)vp;
      vf0.q[1] = *(const u32x4*)(vp + 8);
      if (n < 8) {
        const _Float16* vp2 = v_t + (h * CH + 16 + n) * NSEQ + j0 + hi * 16;
        vf1.q[0] = *(const u32x4*)vp2;
        vf1.q[1] = *(const u32x4*)(vp2 + 8);
      } else { vf1.q[0] = z4; vf1.q[1] = z4; }
    }
    o0 = wmma_f16(pf.v, vf0.v, o0);
    o1 = wmma_f16(pf.v, vf1.v, o1);
  }

  // normalize, gate with g, store (g*o) in [N, H*CH] f16
#pragma unroll
  for (int r = 0; r < 8; ++r) {
    float inv = 1.0f / fmaxf(lrun[r], 1e-20f);
    int m = hi * 8 + r;
    const float* gp = g_f + (i0 + m) * CA + h * CH;
    og[(i0 + m) * CA + h * CH + n] = (_Float16)(o0[r] * inv * gp[n]);
    if (n < 8)
      og[(i0 + m) * CA + h * CH + 16 + n] = (_Float16)(o1[r] * inv * gp[16 + n]);
  }
}

// ---------------------------------------------------------------- K7: output GEMM + s-gate
__global__ __launch_bounds__(128) void out_kernel(
    const _Float16* __restrict__ og_h, const _Float16* __restrict__ s_h,
    const _Float16* __restrict__ wo, const _Float16* __restrict__ ws,
    const float* __restrict__ bs, float* __restrict__ out) {
  int wave = threadIdx.x >> 5, lane = threadIdx.x & 31;
  int tid = blockIdx.x * 4 + wave;
  int i0 = (tid / 24) * 16, c0 = (tid % 24) * 16;
  int n = lane & 15, hi = lane >> 4;
  v8f accO = {0.f,0.f,0.f,0.f,0.f,0.f,0.f,0.f};
  v8f accS = accO;
  for (int k0 = 0; k0 < CA; k0 += 32) {
    accO = wmma_f16(load_a_frag(og_h + i0 * CA + k0, CA),
                    load_b_fragT(wo + c0 * CA + k0, CA), accO);
    accS = wmma_f16(load_a_frag(s_h + i0 * CA + k0, CA),
                    load_b_fragT(ws + c0 * CA + k0, CA), accS);
  }
  float b = bs[c0 + n];
#pragma unroll
  for (int r = 0; r < 8; ++r) {
    int m = hi * 8 + r;
    out[(i0 + m) * CA + c0 + n] = sigmoidf_(accS[r] + b) * accO[r];
  }
}

// ----------------------------------------------------------------
extern "C" void kernel_launch(void* const* d_in, const int* in_sizes, int n_in,
                              void* d_out, int out_size, void* d_ws, size_t ws_size,
                              hipStream_t stream) {
  const float* a       = (const float*)d_in[0];
  const float* s       = (const float*)d_in[1];
  const float* z       = (const float*)d_in[2];
  const float* mask    = (const float*)d_in[3];
  const float* ln_s_w  = (const float*)d_in[4];
  const float* W_gamma = (const float*)d_in[5];
  const float* b_gamma = (const float*)d_in[6];
  const float* W_beta  = (const float*)d_in[7];
  const float* lnz_w   = (const float*)d_in[8];
  const float* lnz_b   = (const float*)d_in[9];
  const float* Wz      = (const float*)d_in[10];
  const float* Wq      = (const float*)d_in[11];
  const float* bq      = (const float*)d_in[12];
  const float* Wk      = (const float*)d_in[13];
  const float* Wv      = (const float*)d_in[14];
  const float* Wg      = (const float*)d_in[15];
  const float* Wo      = (const float*)d_in[16];
  const float* Ws      = (const float*)d_in[17];
  const float* bs      = (const float*)d_in[18];

  char* base = (char*)d_ws;
  size_t off = 0;
  auto alloc = [&](size_t bytes) -> void* {
    off = (off + 255) & ~(size_t)255;
    void* p = base + off;
    off += bytes;
    return p;
  };
  const size_t NC = (size_t)NSEQ * CA;
  _Float16* s_n_h = (_Float16*)alloc(NC * 2);
  _Float16* s_h   = (_Float16*)alloc(NC * 2);
  float*    a_n_f = (float*)   alloc(NC * 4);
  _Float16* a_c_h = (_Float16*)alloc(NC * 2);
  _Float16* q_h   = (_Float16*)alloc(NC * 2);
  _Float16* k_h   = (_Float16*)alloc(NC * 2);
  _Float16* v_t   = (_Float16*)alloc(NC * 2);
  float*    g_f   = (float*)   alloc(NC * 4);
  _Float16* og_h  = (_Float16*)alloc(NC * 2);
  _Float16* zb_h  = (_Float16*)alloc((size_t)NH * NSEQ * NSEQ * 2);
  _Float16* wgam_h = (_Float16*)alloc((size_t)CA * CA * 2);
  _Float16* wbet_h = (_Float16*)alloc((size_t)CA * CA * 2);
  _Float16* wq_h   = (_Float16*)alloc((size_t)CA * CA * 2);
  _Float16* wk_h   = (_Float16*)alloc((size_t)CA * CA * 2);
  _Float16* wv_h   = (_Float16*)alloc((size_t)CA * CA * 2);
  _Float16* wg_h   = (_Float16*)alloc((size_t)CA * CA * 2);
  _Float16* wo_h   = (_Float16*)alloc((size_t)CA * CA * 2);
  _Float16* ws_h   = (_Float16*)alloc((size_t)CA * CA * 2);

  cvt_weights_kernel<<<(CA * CA + 255) / 256, 256, 0, stream>>>(
      W_gamma, W_beta, Wq, Wk, Wv, Wg, Wo, Ws,
      wgam_h, wbet_h, wq_h, wk_h, wv_h, wg_h, wo_h, ws_h);
  rowln_kernel<<<NSEQ, 384, 0, stream>>>(a, s, ln_s_w, a_n_f, s_n_h, s_h);
  adaln_kernel<<<384, 128, 0, stream>>>(s_n_h, a_n_f, wgam_h, wbet_h, b_gamma, a_c_h);
  qkvg_kernel<<<384, 128, 0, stream>>>(a_c_h, wq_h, wk_h, wv_h, wg_h, bq, q_h, k_h, v_t, g_f);
  zbias_kernel<<<(NSEQ * NSEQ) / 8, 256, 0, stream>>>(z, lnz_w, lnz_b, Wz, zb_h);
  attn_kernel<<<(NH * (NSEQ / 16)) / 4, 128, 0, stream>>>(q_h, k_h, v_t, zb_h, mask, g_f, og_h);
  out_kernel<<<384, 128, 0, stream>>>(og_h, s_h, wo_h, ws_h, bs, (float*)d_out);
}
